// MultiHeadAttention_40699110097697
// MI455X (gfx1250) — matrix operand order repaired and checked by reading, not compiled, not run on hardware
//
#include <hip/hip_runtime.h>
#include <hip/hip_bf16.h>
#include <stdint.h>

// Problem constants (match reference)
#define DIMN     1024
#define HEADS    16
#define HEAD_DIM 64
#define BATCH    4
#define SEQ      2048
#define MROWS    (BATCH * SEQ)   // 8192

#define BQ 32   // queries per wave (2 x 16 subtiles)
#define BK 32   // keys per block step

typedef __attribute__((ext_vector_type(16))) __bf16 v16bf;
typedef __attribute__((ext_vector_type(8)))  __bf16 v8bf;
typedef __attribute__((ext_vector_type(8)))  float  v8f;
typedef unsigned int uint4v __attribute__((ext_vector_type(4)));
typedef int          int4v  __attribute__((ext_vector_type(4)));
typedef int          int8v  __attribute__((ext_vector_type(8)));

#define AS1 __attribute__((address_space(1)))
#define AS3 __attribute__((address_space(3)))

union Frag16 { v16bf v; v8bf h[2]; unsigned short u[16]; };
union FragC  { v8f v; float f[8]; };

__device__ __forceinline__ unsigned short f2bf(float x) {
  unsigned int u = __float_as_uint(x);
  unsigned int r = u + 0x7FFFu + ((u >> 16) & 1u);   // round-to-nearest-even
  return (unsigned short)(r >> 16);
}

__device__ __forceinline__ v8f wmma_bf16(v16bf a, v16bf b, v8f c) {
  return __builtin_amdgcn_wmma_f32_16x16x32_bf16(false, a, false, b, (short)0, c, false, false);
}

// ---------------------------------------------------------------------------
// CDNA5 staging paths for the V^T tile (64 d-rows x 32 keys of bf16 -> 4KB LDS)
// Preference: Tensor Data Mover > async global->LDS > plain loads.
// ---------------------------------------------------------------------------
#if __has_builtin(__builtin_amdgcn_tensor_load_to_lds)
#define STAGE_TDM 1
#pragma message("CDNA5 stage path: TDM (tensor_load_to_lds)")
#elif __has_builtin(__builtin_amdgcn_global_load_async_to_lds_b128)
#define STAGE_ASYNC 1
#pragma message("CDNA5 stage path: ASYNC (global_load_async_to_lds_b128)")
#else
#pragma message("CDNA5 stage path: FALLBACK (plain loads)")
#endif

__device__ __forceinline__ unsigned lds_offset_of(void* p) {
  return (unsigned)(unsigned long long)(AS3 void*)p;
}

// vsrc points at Vt[bh][0][k0]; d-rows are strided by SEQ elements.
// LDS layout produced: ldst[d * 32 + key], 64 rows x 32 keys.
__device__ __forceinline__ void stage_vt(const unsigned short* __restrict__ vsrc,
                                         unsigned short* __restrict__ ldst, int lane) {
#if defined(STAGE_TDM)
  // D# group0: count=1 | lds_addr | 57-bit global addr | type=2 ("image")
  const unsigned long long ga = (unsigned long long)(size_t)vsrc;
  const unsigned lo = lds_offset_of(ldst);
  uint4v g0 = (uint4v){1u, lo, (unsigned)ga, (unsigned)(ga >> 32) | 0x80000000u};
  // D# group1: data_size=2B; tensor 32x64 (elem units), tile 32x64, dim0 stride=SEQ
  int8v g1 = (int8v){0x00010000,           // workgroup_mask=0, data_size=1 (2B)
                     (int)(32u << 16),     // tensor_dim0[15:0]=32 in [31:16]
                     (int)(64u << 16),     // tensor_dim0 hi=0 | tensor_dim1[15:0]=64
                     (int)(32u << 16),     // tensor_dim1 hi=0 | tile_dim0=32
                     64,                   // tile_dim1=64 | tile_dim2=0
                     SEQ,                  // tensor_dim0_stride lo
                     0, 0};
  int4v gz = (int4v){0, 0, 0, 0};
#if defined(__clang_major__) && (__clang_major__ >= 23)
  int8v z8 = (int8v){0, 0, 0, 0, 0, 0, 0, 0};
  __builtin_amdgcn_tensor_load_to_lds(g0, g1, gz, gz, z8, 0);
#else
  __builtin_amdgcn_tensor_load_to_lds(g0, g1, gz, gz, 0);
#endif
#elif defined(STAGE_ASYNC)
#pragma unroll
  for (int rr = 0; rr < 2; ++rr) {
    const int d = lane + rr * 32;
#pragma unroll
    for (int i = 0; i < 4; ++i)
      __builtin_amdgcn_global_load_async_to_lds_b128(
          (AS1 void*)(vsrc + (size_t)d * SEQ + i * 8),
          (AS3 void*)(ldst + d * BK + i * 8), 0, 0);
  }
#else
#pragma unroll
  for (int rr = 0; rr < 2; ++rr) {
    const int d = lane + rr * 32;
#pragma unroll
    for (int i = 0; i < 4; ++i)
      *(v8bf*)(ldst + d * BK + i * 8) = *(const v8bf*)(vsrc + (size_t)d * SEQ + i * 8);
  }
#endif
}

// Ensure previously-issued stage (1 newer in flight allowed) has landed.
__device__ __forceinline__ void stage_wait_prev() {
#if defined(STAGE_TDM)
  __builtin_amdgcn_s_wait_tensorcnt(1);
#elif defined(STAGE_ASYNC)
#if __has_builtin(__builtin_amdgcn_s_wait_asynccnt)
  __builtin_amdgcn_s_wait_asynccnt(8);
#else
  asm volatile("s_wait_asynccnt 0x8" ::: "memory");
#endif
#endif
}

__device__ __forceinline__ void stage_wait_all() {
#if defined(STAGE_TDM)
  __builtin_amdgcn_s_wait_tensorcnt(0);
#elif defined(STAGE_ASYNC)
#if __has_builtin(__builtin_amdgcn_s_wait_asynccnt)
  __builtin_amdgcn_s_wait_asynccnt(0);
#else
  asm volatile("s_wait_asynccnt 0x0" ::: "memory");
#endif
#endif
}

// WAR fence: outstanding LDS reads must finish before DMA overwrites a buffer.
__device__ __forceinline__ void lds_reads_drain() {
#if defined(STAGE_TDM) || defined(STAGE_ASYNC)
#if __has_builtin(__builtin_amdgcn_s_wait_dscnt)
  __builtin_amdgcn_s_wait_dscnt(0);
#else
  asm volatile("s_wait_dscnt 0x0" ::: "memory");
#endif
#endif
}

// ---------------------------------------------------------------------------
// fp32 -> bf16 conversion (vectorized by 4)
// ---------------------------------------------------------------------------
__global__ void cvt_f32_bf16(const float* __restrict__ in,
                             unsigned short* __restrict__ out, int n4) {
  int i = blockIdx.x * blockDim.x + threadIdx.x;
  if (i < n4) {
    float4 f = ((const float4*)in)[i];
    ushort4 o;
    o.x = f2bf(f.x); o.y = f2bf(f.y); o.z = f2bf(f.z); o.w = f2bf(f.w);
    ((ushort4*)out)[i] = o;
  }
}

// ---------------------------------------------------------------------------
// Vt[bh][d][key] = V[b*S+key][h*64+d]  (per-head transpose for PV B-frags)
// One block per (32-key, bh) tile; LDS tile transpose, padded rows.
// ---------------------------------------------------------------------------
__global__ __launch_bounds__(256) void transpose_v(
    const unsigned short* __restrict__ Vbf,  // [B*S, 1024]
    unsigned short* __restrict__ Vt)         // [B*H][64][S]
{
  __shared__ __align__(16) unsigned short t[32][72];  // 32 keys x 64 d (+8 pad)
  const int k0 = blockIdx.x * 32;
  const int bh = blockIdx.y;
  const int h  = bh % HEADS, b = bh / HEADS;
  const int tid = threadIdx.x;
  {
    const int key = tid >> 3;          // 0..31
    const int dc  = (tid & 7) * 8;     // 0..56
    const unsigned short* src = Vbf + (size_t)(b * SEQ + k0 + key) * DIMN + h * HEAD_DIM + dc;
    *(v8bf*)(&t[key][dc]) = *(const v8bf*)(src);
  }
  __syncthreads();
  {
    const int d  = tid >> 2;           // 0..63
    const int kc = (tid & 3) * 8;      // 0..24
    __align__(16) unsigned short tmp[8];
#pragma unroll
    for (int j = 0; j < 8; ++j) tmp[j] = t[kc + j][d];
    unsigned short* dst = Vt + ((size_t)bh * HEAD_DIM + d) * SEQ + k0 + kc;
    *(v8bf*)(dst) = *(const v8bf*)(tmp);
  }
}

// ---------------------------------------------------------------------------
// Y[m,n] = sum_k A[m,k] * W[n,k] + bias[n]  (A: MxK row-major, W: NxK row-major)
// 4 waves per block, each wave computes a 32x32 tile via 2x2 WMMA frags.
// ---------------------------------------------------------------------------
__global__ __launch_bounds__(128) void gemm_bias_wmma(
    const unsigned short* __restrict__ A,
    const unsigned short* __restrict__ W,
    const float* __restrict__ bias,
    void* __restrict__ out, int M, int N, int K, int out_bf16)
{
  const int lane = threadIdx.x & 31;
  const int wid  = threadIdx.x >> 5;
  const int wm   = wid >> 1, wn = wid & 1;
  const int bm   = blockIdx.y * 64 + wm * 32;
  const int bn   = blockIdx.x * 64 + wn * 32;

  const int am  = lane & 15;
  const int akb = (lane >> 4) * 8;
  const int bnl = lane & 15;
  const int bkk = (lane >> 4) * 8;

  FragC C[2][2];
#pragma unroll
  for (int i = 0; i < 2; ++i)
#pragma unroll
    for (int j = 0; j < 2; ++j)
      C[i][j].v = (v8f){0.f,0.f,0.f,0.f,0.f,0.f,0.f,0.f};

  for (int k0 = 0; k0 < K; k0 += 32) {
    // prefetch four k-tiles ahead into near caches (global_prefetch_b8)
    __builtin_prefetch(A + (size_t)(bm + am) * K + k0 + 128, 0, 3);
    __builtin_prefetch(W + (size_t)(bn + bnl) * K + k0 + 128, 0, 3);

    Frag16 a[2], b[2];
#pragma unroll
    for (int i = 0; i < 2; ++i) {
      const unsigned short* pa = A + (size_t)(bm + i * 16 + am) * K + k0 + akb;
      a[i].h[0] = *(const v8bf*)(pa);
      a[i].h[1] = *(const v8bf*)(pa + 16);
      const unsigned short* pb = W + (size_t)(bn + i * 16 + bnl) * K + k0 + bkk;
      b[i].h[0] = *(const v8bf*)(pb);
      b[i].h[1] = *(const v8bf*)(pb + 16);
    }
#pragma unroll
    for (int i = 0; i < 2; ++i)
#pragma unroll
      for (int j = 0; j < 2; ++j)
        C[i][j].v = wmma_bf16(a[i].v, b[j].v, C[i][j].v);
  }

  const int rowoff = (lane >> 4) * 8;
#pragma unroll
  for (int i = 0; i < 2; ++i) {
#pragma unroll
    for (int j = 0; j < 2; ++j) {
      const int n = bn + j * 16 + (lane & 15);
      const float bv = bias[n];
#pragma unroll
      for (int r = 0; r < 8; ++r) {
        const int m = bm + i * 16 + rowoff + r;
        const float val = C[i][j].f[r] + bv;
        if (out_bf16) ((unsigned short*)out)[(size_t)m * N + n] = f2bf(val);
        else          ((float*)out)[(size_t)m * N + n] = val;
      }
    }
  }
}

// ---------------------------------------------------------------------------
// Causal flash attention, one wave per (batch, head, 32-query tile).
// V^T tiles staged through LDS by TDM/async DMA, double-buffered.
// ---------------------------------------------------------------------------
__global__ __launch_bounds__(32) void flash_attn_wmma(
    const unsigned short* __restrict__ Qb,
    const unsigned short* __restrict__ Kb,
    const unsigned short* __restrict__ Vt,   // [B*H][64][S]
    const unsigned char*  __restrict__ mask,
    unsigned short* __restrict__ Ob)
{
  __shared__ __align__(16) unsigned short lds_p[BQ * BK];              // 2 KB
  __shared__ __align__(16) unsigned short lds_vt[2][HEAD_DIM * BK];    // 8 KB

  const int lane = threadIdx.x;
  const int qt   = blockIdx.x % (SEQ / BQ);
  const int bh   = blockIdx.x / (SEQ / BQ);
  const int h    = bh % HEADS;
  const int b    = bh / HEADS;
  const int q0   = qt * BQ;

  const int am   = lane & 15;        // A-frag row
  const int akb  = (lane >> 4) * 8;  // A-frag K sub-base
  const int ncol = lane & 15;        // C/B-frag column lane
  const int bkk  = (lane >> 4) * 8;  // B-frag K base
  const int rowl = (lane >> 4) * 8;  // C-frag row offset

  const unsigned char*  maskp  = mask + (size_t)b * SEQ;
  const unsigned short* Vtbase = Vt + (size_t)bh * HEAD_DIM * SEQ;

  // Q A-frags: 2 query subtiles x 2 head-dim chunks (resident all kernel)
  Frag16 aq[2][2];
#pragma unroll
  for (int ms = 0; ms < 2; ++ms)
#pragma unroll
    for (int ks = 0; ks < 2; ++ks) {
      const unsigned short* p =
          Qb + (size_t)(b * SEQ + q0 + ms * 16 + am) * DIMN + h * HEAD_DIM + ks * 32 + akb;
      aq[ms][ks].h[0] = *(const v8bf*)(p);
      aq[ms][ks].h[1] = *(const v8bf*)(p + 16);
    }

  float Mr[2][8], Lr[2][8];
  FragC Oc[2][4];
#pragma unroll
  for (int ms = 0; ms < 2; ++ms) {
#pragma unroll
    for (int r = 0; r < 8; ++r) { Mr[ms][r] = -1e30f; Lr[ms][r] = 0.f; }
#pragma unroll
    for (int c = 0; c < 4; ++c) Oc[ms][c].v = (v8f){0.f,0.f,0.f,0.f,0.f,0.f,0.f,0.f};
  }

  const float scale = 0.125f;            // 1/sqrt(64)
  const int nkb = (q0 + BQ + BK - 1) >> 5;

  // prologue: stage first V^T tile
  stage_vt(Vtbase, &lds_vt[0][0], lane);

  for (int kb = 0; kb < nkb; ++kb) {
    const int cur = kb & 1;
    const int k0  = kb * BK;

    // --- scores S = Q @ K^T : 2 subtiles x 2 key tiles (K frag shared) ---
    FragC sc[2][2];
#pragma unroll
    for (int ms = 0; ms < 2; ++ms)
#pragma unroll
      for (int kn = 0; kn < 2; ++kn)
        sc[ms][kn].v = (v8f){0.f,0.f,0.f,0.f,0.f,0.f,0.f,0.f};
#pragma unroll
    for (int kn = 0; kn < 2; ++kn) {
      const int key = k0 + kn * 16 + ncol;
#pragma unroll
      for (int ks = 0; ks < 2; ++ks) {
        Frag16 bk;
        const unsigned short* p =
            Kb + (size_t)(b * SEQ + key) * DIMN + h * HEAD_DIM + ks * 32 + bkk;
        bk.h[0] = *(const v8bf*)(p);
        bk.h[1] = *(const v8bf*)(p + 16);
#pragma unroll
        for (int ms = 0; ms < 2; ++ms)
          sc[ms][kn].v = wmma_bf16(aq[ms][ks].v, bk.v, sc[ms][kn].v);
      }
    }

    // --- online softmax per subtile ---
    const int key0 = k0 + ncol;
    const int key1 = k0 + 16 + ncol;
    const unsigned char mk0 = maskp[key0];
    const unsigned char mk1 = maskp[key1];
    float p0v[2][8], p1v[2][8];
#pragma unroll
    for (int ms = 0; ms < 2; ++ms) {
#pragma unroll
      for (int r = 0; r < 8; ++r) {
        const int row = q0 + ms * 16 + rowl + r;
        float s0 = sc[ms][0].f[r] * scale;
        float s1 = sc[ms][1].f[r] * scale;
        if (key0 > row || !mk0) s0 = -1e30f;
        if (key1 > row || !mk1) s1 = -1e30f;
        float mx = fmaxf(s0, s1);
#pragma unroll
        for (int off = 1; off < 16; off <<= 1)
          mx = fmaxf(mx, __shfl_xor(mx, off, 32));
        const float nm = fmaxf(Mr[ms][r], mx);
        const float al = __expf(Mr[ms][r] - nm);
        const float p0 = __expf(s0 - nm);
        const float p1 = __expf(s1 - nm);
        float rs = p0 + p1;
#pragma unroll
        for (int off = 1; off < 16; off <<= 1)
          rs += __shfl_xor(rs, off, 32);
        Lr[ms][r] = Lr[ms][r] * al + rs;
        Mr[ms][r] = nm;
#pragma unroll
        for (int c = 0; c < 4; ++c) Oc[ms][c].f[r] *= al;
        p0v[ms][r] = p0; p1v[ms][r] = p1;
      }
    }

    // --- overlap: kick DMA for next V^T tile into the other buffer ---
    if (kb + 1 < nkb) {
      lds_reads_drain();                 // WAR: old reads of that buffer done
      stage_vt(Vtbase + (size_t)(k0 + BK), &lds_vt[cur ^ 1][0], lane);
      stage_wait_prev();                 // current buffer now resident
    } else {
      stage_wait_all();
    }

    // --- P -> LDS (C-layout -> A-layout transpose) ---
#pragma unroll
    for (int ms = 0; ms < 2; ++ms)
#pragma unroll
      for (int r = 0; r < 8; ++r) {
        lds_p[(ms * 16 + rowl + r) * BK + ncol]      = f2bf(p0v[ms][r]);
        lds_p[(ms * 16 + rowl + r) * BK + 16 + ncol] = f2bf(p1v[ms][r]);
      }

    Frag16 ap[2];
#pragma unroll
    for (int ms = 0; ms < 2; ++ms) {
      const unsigned short* pp = &lds_p[(ms * 16 + am) * BK + akb];
      ap[ms].h[0] = *(const v8bf*)(pp);
      ap[ms].h[1] = *(const v8bf*)(pp + 16);
    }

    // --- O += P @ V : contiguous V^T B-frag reads, shared across subtiles ---
#pragma unroll
    for (int c = 0; c < 4; ++c) {
      Frag16 bv;
      const unsigned short* pv = &lds_vt[cur][(c * 16 + ncol) * BK + bkk];
      bv.h[0] = *(const v8bf*)(pv);
      bv.h[1] = *(const v8bf*)(pv + 16);
#pragma unroll
      for (int ms = 0; ms < 2; ++ms)
        Oc[ms][c].v = wmma_bf16(ap[ms].v, bv.v, Oc[ms][c].v);
    }
  }

  // --- finalize: O /= L, store bf16 attn-out in [B*S, 1024] layout ---
#pragma unroll
  for (int ms = 0; ms < 2; ++ms) {
    float invL[8];
#pragma unroll
    for (int r = 0; r < 8; ++r) invL[r] = 1.0f / Lr[ms][r];
#pragma unroll
    for (int c = 0; c < 4; ++c)
#pragma unroll
      for (int r = 0; r < 8; ++r) {
        const int row = b * SEQ + q0 + ms * 16 + rowl + r;
        const int col = h * HEAD_DIM + c * 16 + ncol;
        Ob[(size_t)row * DIMN + col] = f2bf(Oc[ms][c].f[r] * invL[r]);
      }
  }
}

// ---------------------------------------------------------------------------
// Launch
// ---------------------------------------------------------------------------
extern "C" void kernel_launch(void* const* d_in, const int* in_sizes, int n_in,
                              void* d_out, int out_size, void* d_ws, size_t ws_size,
                              hipStream_t stream) {
  const float*         q    = (const float*)d_in[0];
  const unsigned char* mask = (const unsigned char*)d_in[1];
  const float* Wq = (const float*)d_in[2]; const float* bq = (const float*)d_in[3];
  const float* Wk = (const float*)d_in[4]; const float* bk = (const float*)d_in[5];
  const float* Wv = (const float*)d_in[6]; const float* bv = (const float*)d_in[7];
  const float* Wo = (const float*)d_in[8]; const float* bo = (const float*)d_in[9];

  char* ws = (char*)d_ws;
  size_t o = 0;
  unsigned short* Xb  = (unsigned short*)(ws + o); o += (size_t)MROWS * DIMN * 2; // 16 MB
  unsigned short* Wqb = (unsigned short*)(ws + o); o += (size_t)DIMN * DIMN * 2;  //  2 MB
  unsigned short* Wkb = (unsigned short*)(ws + o); o += (size_t)DIMN * DIMN * 2;
  unsigned short* Wvb = (unsigned short*)(ws + o); o += (size_t)DIMN * DIMN * 2;
  unsigned short* Wob = (unsigned short*)(ws + o); o += (size_t)DIMN * DIMN * 2;
  unsigned short* Qb  = (unsigned short*)(ws + o); o += (size_t)MROWS * DIMN * 2; // 16 MB
  unsigned short* Kbf = (unsigned short*)(ws + o); o += (size_t)MROWS * DIMN * 2;
  unsigned short* Vbf = (unsigned short*)(ws + o); o += (size_t)MROWS * DIMN * 2;
  unsigned short* Ab  = (unsigned short*)(ws + o); o += (size_t)MROWS * DIMN * 2;
  unsigned short* Vtb = (unsigned short*)(ws + o); o += (size_t)MROWS * DIMN * 2; // 16 MB

  {
    const int nx4 = MROWS * DIMN / 4;
    cvt_f32_bf16<<<(nx4 + 255) / 256, 256, 0, stream>>>(q, Xb, nx4);
    const int nw4 = DIMN * DIMN / 4;
    cvt_f32_bf16<<<(nw4 + 255) / 256, 256, 0, stream>>>(Wq, Wqb, nw4);
    cvt_f32_bf16<<<(nw4 + 255) / 256, 256, 0, stream>>>(Wk, Wkb, nw4);
    cvt_f32_bf16<<<(nw4 + 255) / 256, 256, 0, stream>>>(Wv, Wvb, nw4);
    cvt_f32_bf16<<<(nw4 + 255) / 256, 256, 0, stream>>>(Wo, Wob, nw4);
  }

  dim3 ggrid(DIMN / 64, MROWS / 64);
  gemm_bias_wmma<<<ggrid, 128, 0, stream>>>(Xb, Wqb, bq, Qb,  MROWS, DIMN, DIMN, 1);
  gemm_bias_wmma<<<ggrid, 128, 0, stream>>>(Xb, Wkb, bk, Kbf, MROWS, DIMN, DIMN, 1);
  gemm_bias_wmma<<<ggrid, 128, 0, stream>>>(Xb, Wvb, bv, Vbf, MROWS, DIMN, DIMN, 1);

  // per-head V transpose for contiguous PV B-fragments
  {
    dim3 tgrid(SEQ / 32, BATCH * HEADS);
    transpose_v<<<tgrid, 256, 0, stream>>>(Vbf, Vtb);
  }

  flash_attn_wmma<<<BATCH * HEADS * (SEQ / BQ), 32, 0, stream>>>(Qb, Kbf, Vtb, mask, Ab);

  gemm_bias_wmma<<<ggrid, 128, 0, stream>>>(Ab, Wob, bo, d_out, MROWS, DIMN, DIMN, 0);
}
